// MultiHeadSelfAttention_81037442941492
// MI455X (gfx1250) — matrix operand order repaired and checked by reading, not compiled, not run on hardware
//
#include <hip/hip_runtime.h>
#include <hip/hip_bf16.h>
#include <math.h>

#ifndef __has_builtin
#define __has_builtin(x) 0
#endif
#if __has_builtin(__builtin_amdgcn_tensor_load_to_lds)
#define HAVE_TDM 1
#else
#define HAVE_TDM 0
#endif

// ---------- types ----------
typedef __bf16 bf16_t;
typedef __bf16 v16bf __attribute__((ext_vector_type(16)));
typedef float  v8f   __attribute__((ext_vector_type(8)));
typedef unsigned int u32x4v __attribute__((ext_vector_type(4)));
typedef int i32x4v __attribute__((ext_vector_type(4)));
typedef int i32x8v __attribute__((ext_vector_type(8)));

union FragU { v16bf bf; u32x4v u[2]; };

__device__ inline v8f vzero8() {
  v8f z;
#pragma unroll
  for (int i = 0; i < 8; ++i) z[i] = 0.0f;
  return z;
}

// A-matrix 16x32 bf16 fragment (M x K), row-major source, ld in elements.
// lane<16:  M=lane,    K chunks [k0,k0+8) and [k0+16,k0+24)
// lane>=16: M=lane-16, K chunks [k0+8,k0+16) and [k0+24,k0+32)
__device__ inline v16bf load_fragA(const bf16_t* base, int ld, int row0, int k0, int lane) {
  const int m = lane & 15, h = lane >> 4;
  const bf16_t* p = base + (size_t)(row0 + m) * ld + k0 + h * 8;
  FragU f;
  f.u[0] = *(const u32x4v*)(p);
  f.u[1] = *(const u32x4v*)(p + 16);
  return f.bf;
}

// B-matrix 32x16 bf16 fragment (K x N); source stored so that for fixed n the K
// values are contiguous: elem(k,n) = base[n*ld + k].
// lane<16:  N=lane,    K chunks [k0,k0+8) and [k0+16,k0+24)
// lane>=16: N=lane-16, K chunks [k0+8,k0+16) and [k0+24,k0+32)
__device__ inline v16bf load_fragB(const bf16_t* base, int ld, int n0, int k0, int lane) {
  const int n = lane & 15, h = lane >> 4;
  const bf16_t* p = base + (size_t)(n0 + n) * ld + k0 + h * 8;
  FragU f;
  f.u[0] = *(const u32x4v*)(p);
  f.u[1] = *(const u32x4v*)(p + 16);
  return f.bf;
}

__device__ inline v8f wmma_bf16(v16bf a, v16bf b, v8f c) {
  return __builtin_amdgcn_wmma_f32_16x16x32_bf16(false, a, false, b, (short)0, c, false, false);
}

// ---------- TDM: 2D tile load Global -> LDS ----------
// Builds a D# (groups 0/1 per CDNA5 ISA 8.3/8.4) and issues TENSOR_LOAD_TO_LDS.
// dims/strides in elements (data_size = 2 bytes). Tile stored compacted
// row-major in LDS (row length = tile_d0 elements).
__device__ inline void tdm_load_2d(unsigned lds_off, const bf16_t* g,
                                   unsigned tensor_d0, unsigned tensor_d1,
                                   unsigned tile_d0, unsigned tile_d1,
                                   unsigned stride0) {
#if HAVE_TDM
  const unsigned long long ga = (unsigned long long)(uintptr_t)g;
  u32x4v g0;
  g0[0] = 1u;                                   // count=1, user descriptor
  g0[1] = lds_off;                              // lds_addr (bytes)
  g0[2] = (unsigned)(ga & 0xFFFFFFFFu);         // global_addr[31:0]
  g0[3] = ((unsigned)(ga >> 32) & 0x01FFFFFFu)  // global_addr[56:32]
          | (2u << 30);                         // type=2 ("image")
  i32x8v g1;
  g1[0] = (int)(1u << 16);                      // data_size=1 (2 bytes)
  g1[1] = (int)((tensor_d0 & 0xFFFFu) << 16);   // tensor_dim0[15:0]
  g1[2] = (int)((tensor_d0 >> 16) | ((tensor_d1 & 0xFFFFu) << 16));
  g1[3] = (int)((tensor_d1 >> 16) | (tile_d0 << 16));
  g1[4] = (int)tile_d1;                         // tile_dim1 (tile_dim2=0)
  g1[5] = (int)stride0;                         // tensor_dim0_stride[31:0]
  g1[6] = 0;
  g1[7] = 0;
  i32x4v z4;
  z4[0] = 0; z4[1] = 0; z4[2] = 0; z4[3] = 0;
#if defined(__clang_major__) && (__clang_major__ >= 23)
  i32x8v z8;
#pragma unroll
  for (int i = 0; i < 8; ++i) z8[i] = 0;
  __builtin_amdgcn_tensor_load_to_lds(g0, g1, z4, z4, z8, 0);
#else
  __builtin_amdgcn_tensor_load_to_lds(g0, g1, z4, z4, 0);
#endif
#else
  (void)lds_off; (void)g; (void)tensor_d0; (void)tensor_d1;
  (void)tile_d0; (void)tile_d1; (void)stride0;
#endif
}

// Fallback cooperative copy (only used if the TDM builtin is unavailable).
__device__ inline void stage_copy_fallback(bf16_t* dst, const bf16_t* src,
                                           int rows, int srcld, int tid) {
  const int chunks = rows * 8;  // 64 bf16 per row = 8 x 16B
  for (int c = tid; c < chunks; c += 128) {
    const int r = c >> 3, q = c & 7;
    *(u32x4v*)(dst + r * 64 + q * 8) = *(const u32x4v*)(src + (size_t)r * srcld + q * 8);
  }
}

// ---------- conversion kernels ----------
__global__ void cvt_f32_to_bf16(const float* __restrict__ src, bf16_t* __restrict__ dst, int n) {
  int i = (blockIdx.x * blockDim.x + threadIdx.x) * 4;
  if (i < n) {
    float4 v = *(const float4*)(src + i);
    dst[i + 0] = (bf16_t)v.x;
    dst[i + 1] = (bf16_t)v.y;
    dst[i + 2] = (bf16_t)v.z;
    dst[i + 3] = (bf16_t)v.w;
  }
}

// (H,64,32) f32 -> (H,32,64) bf16
__global__ void cvt_lsr_transpose(const float* __restrict__ src, bf16_t* __restrict__ dst) {
  int i = blockIdx.x * blockDim.x + threadIdx.x;
  if (i < 16 * 64 * 32) {
    int h = i >> 11;
    int d = (i >> 5) & 63;
    int r = i & 31;
    dst[(h << 11) + (r << 6) + d] = (bf16_t)src[i];
  }
}

// ---------- main GEMM: C = A(MxK) * B^T(NxK row-major) + bias ----------
// TDM double-buffered LDS staging: block tile 128x64, K staged in 64-chunks.
// MODE 0: store bf16 row-major (M,N)
// MODE 1: store bf16 transposed per head: (B,H,64,T), n=h*64+d, m=b*2048+t
// MODE 2: store f32 row-major (M,N)
template <int MODE>
__global__ __launch_bounds__(128)
void gemm_bf16_wmma(const bf16_t* __restrict__ A, const bf16_t* __restrict__ Bw,
                    const float* __restrict__ bias, void* __restrict__ Out,
                    int M, int N, int K) {
  // [buf][ A: 128x64 | B: 64x64 ]  => 2*(8192+4096)*2B = 48 KB
  __shared__ __attribute__((aligned(128))) bf16_t lds[2][128 * 64 + 64 * 64];
  const int tid = threadIdx.x;
  const int lane = tid & 31;
  const int wave = tid >> 5;
  const int mblk = blockIdx.x * 128;
  const int nblk = blockIdx.y * 64;
  const int m0 = (wave >> 1) * 64;  // local wave tile origin
  const int n0 = (wave & 1) * 32;
  const int nk = K >> 6;  // K/64 stages

  auto stage = [&](int kt, int buf) {
    const bf16_t* ga = A + (size_t)mblk * K + kt * 64;
    const bf16_t* gb = Bw + (size_t)nblk * K + kt * 64;
#if HAVE_TDM
    if (wave == 0) {
      const unsigned offA = (unsigned)(uintptr_t)&lds[buf][0];
      const unsigned offB = (unsigned)(uintptr_t)&lds[buf][128 * 64];
      tdm_load_2d(offA, ga, (unsigned)K, (unsigned)M, 64u, 128u, (unsigned)K);
      tdm_load_2d(offB, gb, (unsigned)K, (unsigned)N, 64u, 64u, (unsigned)K);
    }
#else
    stage_copy_fallback(&lds[buf][0], ga, 128, K, tid);
    stage_copy_fallback(&lds[buf][128 * 64], gb, 64, K, tid);
#endif
  };

  v8f acc[4][2];
#pragma unroll
  for (int i = 0; i < 4; ++i)
#pragma unroll
    for (int j = 0; j < 2; ++j) acc[i][j] = vzero8();

  stage(0, 0);
  for (int kt = 0; kt < nk; ++kt) {
    const int buf = kt & 1;
#if HAVE_TDM
    if (wave == 0) __builtin_amdgcn_s_wait_tensorcnt(0);
#endif
    __syncthreads();  // current buffer resident + previous reads of other buffer done
    if (kt + 1 < nk) stage(kt + 1, buf ^ 1);

    const bf16_t* pA = &lds[buf][0];
    const bf16_t* pB = &lds[buf][128 * 64];
#pragma unroll
    for (int ks = 0; ks < 64; ks += 32) {
      v16bf af[4], bfr[2];
#pragma unroll
      for (int i = 0; i < 4; ++i) af[i] = load_fragA(pA, 64, m0 + 16 * i, ks, lane);
#pragma unroll
      for (int j = 0; j < 2; ++j) bfr[j] = load_fragB(pB, 64, n0 + 16 * j, ks, lane);
#pragma unroll
      for (int i = 0; i < 4; ++i)
#pragma unroll
        for (int j = 0; j < 2; ++j) acc[i][j] = wmma_bf16(af[i], bfr[j], acc[i][j]);
    }
  }

  const int cn = lane & 15, chh = lane >> 4;
#pragma unroll
  for (int i = 0; i < 4; ++i) {
#pragma unroll
    for (int j = 0; j < 2; ++j) {
      const int n = nblk + n0 + j * 16 + cn;
      const float bv = bias[n];
#pragma unroll
      for (int r = 0; r < 8; ++r) {
        const int m = mblk + m0 + i * 16 + chh * 8 + r;
        const float val = acc[i][j][r] + bv;
        if (MODE == 0) {
          ((bf16_t*)Out)[(size_t)m * N + n] = (bf16_t)val;
        } else if (MODE == 1) {
          const int bb = m >> 11, t = m & 2047, hh = n >> 6, d = n & 63;
          ((bf16_t*)Out)[((((size_t)bb * 16 + hh) * 64) + d) * 2048 + t] = (bf16_t)val;
        } else {
          ((float*)Out)[(size_t)m * N + n] = val;
        }
      }
    }
  }
}

// ---------- per-head low-rank projection: (T,64) @ (64,32) -> (T,32) bf16 ----------
__global__ __launch_bounds__(128)
void lsr_proj_wmma(const bf16_t* __restrict__ QK,  // (B*T, 1024) bf16
                   const bf16_t* __restrict__ Wt,  // (H, 32, 64) bf16 (K-contig per rank)
                   bf16_t* __restrict__ out,       // (B*H, T, 32) bf16
                   float scale) {
  const int lane = threadIdx.x & 31, wave = threadIdx.x >> 5;
  const int bh = blockIdx.x;
  const int b = bh >> 4, h = bh & 15;
  const int t0 = blockIdx.y * 64 + wave * 16;
  const bf16_t* Ab = QK + ((size_t)b * 2048) * 1024 + h * 64;  // row t, col d, ld 1024
  const bf16_t* Bb = Wt + (size_t)h * 32 * 64;                 // row r,  col d, ld 64

  v8f acc[2] = {vzero8(), vzero8()};
#pragma unroll
  for (int k0 = 0; k0 < 64; k0 += 32) {
    v16bf a = load_fragA(Ab, 1024, t0, k0, lane);
#pragma unroll
    for (int j = 0; j < 2; ++j) {
      v16bf bb = load_fragB(Bb, 64, j * 16, k0, lane);
      acc[j] = wmma_bf16(a, bb, acc[j]);
    }
  }
  const int cn = lane & 15, chh = lane >> 4;
#pragma unroll
  for (int j = 0; j < 2; ++j) {
    const int r = j * 16 + cn;
#pragma unroll
    for (int rr = 0; rr < 8; ++rr) {
      const int t = t0 + chh * 8 + rr;
      out[((size_t)bh * 2048 + t) * 32 + r] = (bf16_t)(acc[j][rr] * scale);
    }
  }
}

// ---------- flash attention (causal, low-rank scores) ----------
// qlr/klr: (B*H, T, 32) bf16 (scale folded into qlr); vt: (B*H, 64, T) bf16
// yb: (B*T, 1024) bf16
__global__ __launch_bounds__(128)
void flash_attn_wmma(const bf16_t* __restrict__ qlr, const bf16_t* __restrict__ klr,
                     const bf16_t* __restrict__ vt, bf16_t* __restrict__ yb) {
  __shared__ __attribute__((aligned(16))) bf16_t ldsP[4][16 * 32];
  const int lane = threadIdx.x & 31, wave = threadIdx.x >> 5;
  const int bh = blockIdx.x;
  const int b = bh >> 4, h = bh & 15;
  const int q0 = blockIdx.y * 64 + wave * 16;  // query row base (within T)
  const bf16_t* qb = qlr + (size_t)bh * 2048 * 32;
  const bf16_t* kb = klr + (size_t)bh * 2048 * 32;
  const bf16_t* vb = vt + (size_t)bh * 64 * 2048;

  const v16bf aq = load_fragA(qb, 32, q0, 0, lane);  // 16 queries x full rank 32
  v8f y[4];
#pragma unroll
  for (int f = 0; f < 4; ++f) y[f] = vzero8();
  float mi[8], li[8];
#pragma unroll
  for (int r = 0; r < 8; ++r) { mi[r] = -3.0e38f; li[r] = 0.0f; }

  const int cn = lane & 15, chh = lane >> 4;
  const int jlast = q0 + 15;  // highest key this tile may attend to

  for (int j0 = 0; j0 <= jlast; j0 += 32) {
    if (j0 + 32 <= jlast)
      __builtin_prefetch(kb + (size_t)(j0 + 32) * 32, 0, 1);
    v16bf bk0 = load_fragB(kb, 32, j0, 0, lane);       // keys j0..j0+15
    v16bf bk1 = load_fragB(kb, 32, j0 + 16, 0, lane);  // keys j0+16..j0+31
    v8f s0 = wmma_bf16(aq, bk0, vzero8());
    v8f s1 = wmma_bf16(aq, bk1, vzero8());
    const int jc0 = j0 + cn, jc1 = j0 + 16 + cn;
#pragma unroll
    for (int r = 0; r < 8; ++r) {
      const int m = q0 + chh * 8 + r;
      float x0 = (jc0 <= m) ? s0[r] : -3.0e38f;  // causal mask
      float x1 = (jc1 <= m) ? s1[r] : -3.0e38f;
      float mx = fmaxf(x0, x1);
#pragma unroll
      for (int dd = 8; dd >= 1; dd >>= 1) mx = fmaxf(mx, __shfl_xor(mx, dd, 32));
      const float mnew = fmaxf(mi[r], mx);
      const float alpha = __expf(mi[r] - mnew);
      const float p0 = __expf(x0 - mnew);
      const float p1 = __expf(x1 - mnew);
      float ps = p0 + p1;
#pragma unroll
      for (int dd = 8; dd >= 1; dd >>= 1) ps += __shfl_xor(ps, dd, 32);
      li[r] = li[r] * alpha + ps;
      mi[r] = mnew;
#pragma unroll
      for (int f = 0; f < 4; ++f) y[f][r] *= alpha;
      const int mrow = chh * 8 + r;
      ldsP[wave][mrow * 32 + cn] = (bf16_t)p0;
      ldsP[wave][mrow * 32 + 16 + cn] = (bf16_t)p1;
    }
    // reload P tile (16x32) in A-layout from LDS (wave-private; dscnt handled by compiler)
    const v16bf ap = load_fragA(&ldsP[wave][0], 32, 0, 0, lane);
#pragma unroll
    for (int f = 0; f < 4; ++f) {
      v16bf bv = load_fragB(vb, 2048, f * 16, j0, lane);  // V^T: (d, t)
      y[f] = wmma_bf16(ap, bv, y[f]);
    }
  }

#pragma unroll
  for (int f = 0; f < 4; ++f) {
    const int d = f * 16 + cn;
#pragma unroll
    for (int r = 0; r < 8; ++r) {
      const int t = q0 + chh * 8 + r;
      const float val = y[f][r] / li[r];
      yb[((size_t)b * 2048 + t) * 1024 + h * 64 + d] = (bf16_t)val;
    }
  }
}

// ---------- host launcher ----------
extern "C" void kernel_launch(void* const* d_in, const int* in_sizes, int n_in,
                              void* d_out, int out_size, void* d_ws, size_t ws_size,
                              hipStream_t stream) {
  (void)in_sizes; (void)n_in; (void)out_size; (void)ws_size;
  const float* x     = (const float*)d_in[0];
  const float* Wq    = (const float*)d_in[1];
  const float* bq    = (const float*)d_in[2];
  const float* Wk    = (const float*)d_in[3];
  const float* bk    = (const float*)d_in[4];
  const float* Wv    = (const float*)d_in[5];
  const float* bv    = (const float*)d_in[6];
  const float* Wo    = (const float*)d_in[7];
  const float* bo    = (const float*)d_in[8];
  const float* Wqlsr = (const float*)d_in[9];
  const float* Wklsr = (const float*)d_in[10];

  const size_t MT = 8192, DM = 1024, BH = 64, T = 2048, R = 32;

  char* w = (char*)d_ws;
  size_t off = 0;
  auto take = [&](size_t bytes) -> char* {
    char* p = w + off;
    off = (off + bytes + 255) & ~(size_t)255;
    return p;
  };
  bf16_t* Xb   = (bf16_t*)take(MT * DM * 2);
  bf16_t* Wqb  = (bf16_t*)take(DM * DM * 2);
  bf16_t* Wkb  = (bf16_t*)take(DM * DM * 2);
  bf16_t* Wvb  = (bf16_t*)take(DM * DM * 2);
  bf16_t* Wob  = (bf16_t*)take(DM * DM * 2);
  bf16_t* Wqlt = (bf16_t*)take(16 * 32 * 64 * 2);
  bf16_t* Wklt = (bf16_t*)take(16 * 32 * 64 * 2);
  bf16_t* Qb   = (bf16_t*)take(MT * DM * 2);
  bf16_t* Kb   = (bf16_t*)take(MT * DM * 2);
  bf16_t* Vt   = (bf16_t*)take(BH * 64 * T * 2);
  bf16_t* Qlr  = (bf16_t*)take(BH * T * R * 2);
  bf16_t* Klr  = (bf16_t*)take(BH * T * R * 2);
  bf16_t* Yb   = (bf16_t*)take(MT * DM * 2);

  // conversions
  {
    int n = (int)(MT * DM);
    cvt_f32_to_bf16<<<(n / 4 + 255) / 256, 256, 0, stream>>>(x, Xb, n);
    n = (int)(DM * DM);
    int g = (n / 4 + 255) / 256;
    cvt_f32_to_bf16<<<g, 256, 0, stream>>>(Wq, Wqb, n);
    cvt_f32_to_bf16<<<g, 256, 0, stream>>>(Wk, Wkb, n);
    cvt_f32_to_bf16<<<g, 256, 0, stream>>>(Wv, Wvb, n);
    cvt_f32_to_bf16<<<g, 256, 0, stream>>>(Wo, Wob, n);
    cvt_lsr_transpose<<<(16 * 64 * 32 + 255) / 256, 256, 0, stream>>>(Wqlsr, Wqlt);
    cvt_lsr_transpose<<<(16 * 64 * 32 + 255) / 256, 256, 0, stream>>>(Wklsr, Wklt);
  }

  dim3 gGemm(64, 16), bGemm(128);
  gemm_bf16_wmma<0><<<gGemm, bGemm, 0, stream>>>(Xb, Wqb, bq, Qb, 8192, 1024, 1024);
  gemm_bf16_wmma<0><<<gGemm, bGemm, 0, stream>>>(Xb, Wkb, bk, Kb, 8192, 1024, 1024);
  gemm_bf16_wmma<1><<<gGemm, bGemm, 0, stream>>>(Xb, Wvb, bv, Vt, 8192, 1024, 1024);

  dim3 gL(64, 32), bL(128);
  lsr_proj_wmma<<<gL, bL, 0, stream>>>(Qb, Wqlt, Qlr, 0.17677669529663687f);  // 1/sqrt(32)
  lsr_proj_wmma<<<gL, bL, 0, stream>>>(Kb, Wklt, Klr, 1.0f);

  flash_attn_wmma<<<dim3(64, 32), 128, 0, stream>>>(Qlr, Klr, Vt, Yb);

  gemm_bf16_wmma<2><<<gGemm, bGemm, 0, stream>>>(Yb, Wob, bo, d_out, 8192, 1024, 1024);
}
